// AttentionHead_1743756722772
// MI455X (gfx1250) — matrix operand order repaired and checked by reading, not compiled, not run on hardware
//
#include <hip/hip_runtime.h>

typedef __attribute__((ext_vector_type(16))) __bf16 v16bf;
typedef __attribute__((ext_vector_type(8)))  float  v8f;
typedef __attribute__((ext_vector_type(4)))  int    v4i;

typedef __attribute__((address_space(1))) v4i as1_v4i;
typedef __attribute__((address_space(3))) v4i as3_v4i;

constexpr int BB = 4;
constexpr int SS = 4096;
constexpr int EE = 1024;
constexpr int DD = 128;
constexpr size_t BSD = (size_t)BB * SS * DD;   // elements per Q/K/V matrix

// ---- CDNA5 async global->LDS copy (ASYNCcnt), with safe fallback ----------
__device__ inline void async_ld16(const __bf16* g, __bf16* l)
{
#if __has_builtin(__builtin_amdgcn_global_load_async_to_lds_b128)
    __builtin_amdgcn_global_load_async_to_lds_b128((as1_v4i*)g, (as3_v4i*)l, 0, 0);
#else
    // synchronous fallback: 16B through registers
    const uint4 t = *(const uint4*)g;
    *(uint4*)l = t;
#endif
}

__device__ inline void wait_async0()
{
#if __has_builtin(__builtin_amdgcn_s_wait_asynccnt)
    __builtin_amdgcn_s_wait_asynccnt(0);
#else
    asm volatile("s_wait_asynccnt 0" ::: "memory");
#endif
}

// ---------------------------------------------------------------------------
// Phase 1: fused QKV projection.  out[mi] = x @ W[mi], stored bf16 [B*S, D].
// grid = ((B*S)/128, 3), block = 256 (8 waves).
// Wave -> 16 rows x 128 cols (8 WMMA accumulators), double-buffered K-steps.
// ---------------------------------------------------------------------------
__global__ __launch_bounds__(256)
void qkv_proj_kernel(const float* __restrict__ x,
                     const float* __restrict__ Wq,
                     const float* __restrict__ Wk,
                     const float* __restrict__ Wv,
                     unsigned short* __restrict__ qkv_raw)
{
    __shared__ __bf16 wt[2][DD][32];            // transposed weight chunk [col][k]

    const int lane  = threadIdx.x & 31;
    const int wave  = threadIdx.x >> 5;
    const int lmod  = lane & 15;
    const int lhalf = lane >> 4;

    const int r0 = blockIdx.x * 128 + wave * 16;          // output rows
    const int mi = blockIdx.y;                            // 0=Q,1=K,2=V
    const float* W = (mi == 0) ? Wq : (mi == 1) ? Wk : Wv;
    __bf16* out = reinterpret_cast<__bf16*>(qkv_raw) + (size_t)mi * BSD;

    const float* xrow = x + (size_t)(r0 + lmod) * EE;

    // staging decomposition: 32 k-rows x 128 cols of W per chunk, 16 f32/thread
    const int krow = threadIdx.x >> 3;                    // 0..31
    const int cc   = (threadIdx.x & 7) * 16;              // 0..112

    float wreg[16], xcur[16], xnxt[16];

    // prologue: stage chunk 0 into buffer 0
    {
        const float* wp = W + (size_t)krow * DD + cc;
        #pragma unroll
        for (int i = 0; i < 16; ++i) wreg[i] = wp[i];
        #pragma unroll
        for (int e = 0; e < 16; ++e)
            xcur[e] = xrow[16 * lhalf + e];
        #pragma unroll
        for (int i = 0; i < 16; ++i) wt[0][cc + i][krow] = (__bf16)wreg[i];
    }
    __syncthreads();

    v8f acc[8];
    const v8f vzero = {};
    #pragma unroll
    for (int j = 0; j < 8; ++j) acc[j] = vzero;

    int pbuf = 0;
    for (int kk = 0; kk < EE; kk += 32) {
        const bool pf = (kk + 32) < EE;
        if (pf) {
            const float* wp = W + (size_t)(kk + 32 + krow) * DD + cc;
            #pragma unroll
            for (int i = 0; i < 16; ++i) wreg[i] = wp[i];
            #pragma unroll
            for (int e = 0; e < 16; ++e)
                xnxt[e] = xrow[kk + 32 + 16 * lhalf + e];
        }

        v16bf a;
        #pragma unroll
        for (int e = 0; e < 16; ++e) a[e] = (__bf16)xcur[e];

        #pragma unroll
        for (int j = 0; j < 8; ++j) {
            v16bf b;
            #pragma unroll
            for (int e = 0; e < 16; ++e)
                b[e] = wt[pbuf][16 * j + lmod][16 * lhalf + e];
            acc[j] = __builtin_amdgcn_wmma_f32_16x16x32_bf16(false, a, false, b,
                                                             (short)0, acc[j], false, false);
        }

        if (pf) {
            #pragma unroll
            for (int i = 0; i < 16; ++i) wt[1 - pbuf][cc + i][krow] = (__bf16)wreg[i];
            #pragma unroll
            for (int e = 0; e < 16; ++e) xcur[e] = xnxt[e];
        }
        __syncthreads();
        pbuf ^= 1;
    }

    #pragma unroll
    for (int j = 0; j < 8; ++j)
        #pragma unroll
        for (int r = 0; r < 8; ++r) {
            const int row = r0 + r + 8 * lhalf;
            out[(size_t)row * DD + 16 * j + lmod] = (__bf16)acc[j][r];
        }
}

// ---------------------------------------------------------------------------
// Phase 2: flash-attention over bf16 Q/K/V.  grid = B*(S/128), block = 256.
// Wave -> 16-row query tile; block shares 32-key K/V chunks in LDS.
// Double-buffered: K chunk via async global->LDS DMA, V chunk via registers
// with a transposed LDS store.  One barrier per chunk.
// ---------------------------------------------------------------------------
__global__ __launch_bounds__(256)
void attn_kernel(const unsigned short* __restrict__ qkv_raw,
                 float* __restrict__ out)
{
    __shared__ __bf16 k_lds[2][32][DD + 8];   // [key][d], row stride 272B
    __shared__ __bf16 vt_lds[2][DD][48];      // [d][key], row stride 96B
    __shared__ __bf16 p_lds[8][16][32];       // per-wave P staging (C->A relayout)

    const int lane  = threadIdx.x & 31;
    const int wave  = threadIdx.x >> 5;
    const int lmod  = lane & 15;
    const int lhalf = lane >> 4;

    const int b  = blockIdx.x >> 5;           // 32 query tiles per batch
    const int qt = blockIdx.x & 31;
    const int q0 = qt * 128 + wave * 16;

    const __bf16* Q = reinterpret_cast<const __bf16*>(qkv_raw) + (size_t)b * SS * DD;
    const __bf16* K = reinterpret_cast<const __bf16*>(qkv_raw) + BSD     + (size_t)b * SS * DD;
    const __bf16* V = reinterpret_cast<const __bf16*>(qkv_raw) + 2 * BSD + (size_t)b * SS * DD;

    // staging decomposition: 32 keys x 128 d per chunk, 16 bf16/thread
    const int krow = threadIdx.x >> 3;        // 0..31
    const int d0   = (threadIdx.x & 7) * 16;  // 0..112

    // Q fragments (A operands), resident for the whole kernel
    v16bf qf[4];
    {
        const __bf16* qrow = Q + (size_t)(q0 + lmod) * DD;
        #pragma unroll
        for (int s2 = 0; s2 < 4; ++s2)
            #pragma unroll
            for (int e = 0; e < 16; ++e)
                qf[s2][e] = qrow[32 * s2 + 16 * lhalf + e];
    }

    float m_run[8], l_run[8];
    v8f o[8];
    const v8f vzero = {};
    #pragma unroll
    for (int r = 0; r < 8; ++r) { m_run[r] = -3.0e38f; l_run[r] = 0.0f; o[r] = vzero; }

    const float scale = 0.088388347648318447f;   // 1/sqrt(128)

    // prologue: stage chunk 0 into buffer 0
    {
        const __bf16* kp = K + (size_t)krow * DD + d0;
        async_ld16(kp,     &k_lds[0][krow][d0]);
        async_ld16(kp + 8, &k_lds[0][krow][d0 + 8]);
        const __bf16* vp = V + (size_t)krow * DD + d0;
        __bf16 vreg[16];
        #pragma unroll
        for (int i = 0; i < 16; ++i) vreg[i] = vp[i];
        #pragma unroll
        for (int i = 0; i < 16; ++i) vt_lds[0][d0 + i][krow] = vreg[i];
        wait_async0();
    }
    __syncthreads();

    int pbuf = 0;
    for (int kc = 0; kc < SS; kc += 32) {
        const bool pf = (kc + 32) < SS;
        __bf16 vreg[16];
        if (pf) {
            const __bf16* kp = K + (size_t)(kc + 32 + krow) * DD + d0;
            async_ld16(kp,     &k_lds[1 - pbuf][krow][d0]);
            async_ld16(kp + 8, &k_lds[1 - pbuf][krow][d0 + 8]);
            const __bf16* vp = V + (size_t)(kc + 32 + krow) * DD + d0;
            #pragma unroll
            for (int i = 0; i < 16; ++i) vreg[i] = vp[i];
        }

        // scores: two 16x16 tiles over this 32-key chunk, K-dim = D = 4 x 32
        v8f s0 = vzero, s1 = vzero;
        #pragma unroll
        for (int s2 = 0; s2 < 4; ++s2) {
            v16bf b0, b1;
            #pragma unroll
            for (int e = 0; e < 16; ++e) {
                const int d = 32 * s2 + 16 * lhalf + e;
                b0[e] = k_lds[pbuf][lmod][d];
                b1[e] = k_lds[pbuf][16 + lmod][d];
            }
            s0 = __builtin_amdgcn_wmma_f32_16x16x32_bf16(false, qf[s2], false, b0,
                                                         (short)0, s0, false, false);
            s1 = __builtin_amdgcn_wmma_f32_16x16x32_bf16(false, qf[s2], false, b1,
                                                         (short)0, s1, false, false);
        }

        // online softmax (per-row reductions across the 16-lane half)
        float alpha[8];
        #pragma unroll
        for (int r = 0; r < 8; ++r) {
            const float a0 = s0[r] * scale;
            const float a1 = s1[r] * scale;
            float mx = fmaxf(a0, a1);
            #pragma unroll
            for (int w = 8; w >= 1; w >>= 1)
                mx = fmaxf(mx, __shfl_xor(mx, w, 32));
            const float mnew = fmaxf(m_run[r], mx);
            alpha[r] = __expf(m_run[r] - mnew);
            m_run[r] = mnew;
            const float p0 = __expf(a0 - mnew);
            const float p1 = __expf(a1 - mnew);
            s0[r] = p0;
            s1[r] = p1;
            float rs = p0 + p1;
            #pragma unroll
            for (int w = 8; w >= 1; w >>= 1)
                rs += __shfl_xor(rs, w, 32);
            l_run[r] = l_run[r] * alpha[r] + rs;
        }

        #pragma unroll
        for (int j = 0; j < 8; ++j)
            #pragma unroll
            for (int r = 0; r < 8; ++r)
                o[j][r] *= alpha[r];

        // P: C layout -> A layout via per-wave LDS (wave-internal: dscnt only)
        #pragma unroll
        for (int r = 0; r < 8; ++r) {
            p_lds[wave][r + 8 * lhalf][lmod]      = (__bf16)s0[r];
            p_lds[wave][r + 8 * lhalf][16 + lmod] = (__bf16)s1[r];
        }
        asm volatile("s_wait_dscnt 0" ::: "memory");

        v16bf pfrag;
        #pragma unroll
        for (int e = 0; e < 16; ++e)
            pfrag[e] = p_lds[wave][lmod][16 * lhalf + e];

        // O += P @ V  (K-dim = 32-key chunk), 8 column tiles of D=128
        #pragma unroll
        for (int j = 0; j < 8; ++j) {
            v16bf vf;
            #pragma unroll
            for (int e = 0; e < 16; ++e)
                vf[e] = vt_lds[pbuf][16 * j + lmod][16 * lhalf + e];
            o[j] = __builtin_amdgcn_wmma_f32_16x16x32_bf16(false, pfrag, false, vf,
                                                           (short)0, o[j], false, false);
        }

        if (pf) {
            wait_async0();                         // K chunk DMA complete (this wave)
            #pragma unroll
            for (int i = 0; i < 16; ++i) vt_lds[1 - pbuf][d0 + i][krow] = vreg[i];
        }
        __syncthreads();                           // next buffer fully staged
        pbuf ^= 1;
    }

    // epilogue: normalize by row sums and store fp32
    float* orow = out + ((size_t)b * SS + q0) * DD;
    #pragma unroll
    for (int r = 0; r < 8; ++r) {
        const float inv = 1.0f / l_run[r];
        const int m = r + 8 * lhalf;
        #pragma unroll
        for (int j = 0; j < 8; ++j)
            orow[(size_t)m * DD + 16 * j + lmod] = o[j][r] * inv;
    }
}

// ---------------------------------------------------------------------------
extern "C" void kernel_launch(void* const* d_in, const int* in_sizes, int n_in,
                              void* d_out, int out_size, void* d_ws, size_t ws_size,
                              hipStream_t stream)
{
    const float* x  = (const float*)d_in[0];
    const float* Wq = (const float*)d_in[1];
    const float* Wk = (const float*)d_in[2];
    const float* Wv = (const float*)d_in[3];
    float* out = (float*)d_out;
    unsigned short* qkv = (unsigned short*)d_ws;   // 3 x 4 MB bf16 Q/K/V

    dim3 g1((BB * SS) / 128, 3);                   // 128 x 3
    qkv_proj_kernel<<<g1, 256, 0, stream>>>(x, Wq, Wk, Wv, qkv);

    dim3 g2(BB * (SS / 128));                      // 128 blocks
    attn_kernel<<<g2, 256, 0, stream>>>(qkv, out);
}
